// Dynamic_conv2d_87806311399735
// MI455X (gfx1250) — matrix operand order repaired and checked by reading, not compiled, not run on hardware
//
#include <hip/hip_runtime.h>

// ---------------------------------------------------------------------------
// Dynamic conv2d (B=16, C=64->64, 128x128, 3x3 pad1) via bf16x3 WMMA implicit
// GEMM on gfx1250. fp32 operands split into bf16 hi/lo (hi*hi + hi*lo + lo*hi,
// fp32 accumulate, v_wmma_f32_16x16x32_bf16). x is staged once per workgroup
// into LDS (pre-split, halo'd, zero-padded) and reused across all 4 C_out
// tiles, 9 taps and 2 output rows; inner loop is ds_load_b128 -> WMMA only.
// ---------------------------------------------------------------------------

typedef __attribute__((ext_vector_type(16))) __bf16       v16bf;
typedef __attribute__((ext_vector_type(8)))  float        v8f;
typedef __attribute__((ext_vector_type(4)))  unsigned int v4u;

#define NB   16
#define CI   64
#define CO   64
#define HH   128
#define WW   128
#define HW   (HH * WW)
#define TAPS 9
#define NK   5
#define NWEL (NB * TAPS * CO * CI)   // 589824 aggregated weight elements

#define ROWS  4                      // staged input rows (2 outputs + halo)
#define COLS  66                     // 64 output cols + halo
#define NSTG  (ROWS * COLS * CI)     // 16896 staged elements per array

union Frag {
    v16bf v;
    v4u   u[2];
};

// ---------------------------------------------------------------------------
// Kernel 1: aggregate the kernel bank per sample, pre-split into bf16 hi/lo,
// layout [b][tap][co][ci] so A-fragments are two contiguous 16B loads/lane.
// ---------------------------------------------------------------------------
__global__ __launch_bounds__(256) void agg_split_kernel(
    const float* __restrict__ weights,   // [B][K]
    const float* __restrict__ Wbank,     // [K][CO][CI][3][3]
    const float* __restrict__ bbank,     // [K][CO]
    __bf16* __restrict__ whi,            // [B][9][CO][CI]
    __bf16* __restrict__ wlo,            // [B][9][CO][CI]
    float*  __restrict__ aggb)           // [B][CO]
{
    const int gid = blockIdx.x * 256 + threadIdx.x;

    if (gid < NB * CO) {                 // bias aggregation piggy-backs here
        const int bb = gid >> 6, co = gid & 63;
        float s = 0.0f;
        #pragma unroll
        for (int k = 0; k < NK; ++k)
            s += weights[bb * NK + k] * bbank[k * CO + co];
        aggb[gid] = s;
    }

    if (gid >= NWEL) return;
    const int ci  = gid & 63;
    const int co  = (gid >> 6) & 63;
    const int bt  = gid >> 12;           // b*9 + tap
    const int tap = bt % TAPS;
    const int bb  = bt / TAPS;

    float s = 0.0f;
    #pragma unroll
    for (int k = 0; k < NK; ++k)
        s += weights[bb * NK + k] *
             Wbank[((k * CO + co) * CI + ci) * TAPS + tap];

    const __bf16 h = (__bf16)s;
    whi[gid] = h;
    wlo[gid] = (__bf16)(s - (float)h);
}

// ---------------------------------------------------------------------------
// Kernel 2: implicit-GEMM conv. Workgroup = 512 threads (16 waves), owns one
// sample strip: 2 output rows x 64 cols x all 64 C_out.
//   wave w: coT = w&3 (16 C_out), pxT = w>>2 (16 pixels)
// Phase 1: stage x[b][*][y0-1..y0+2][xB-1..xB+64] into LDS as bf16 hi/lo,
//          layout [row][col][ci] (ci contiguous -> b128 fragment reads).
// Phase 2: 9 taps x 2 ci-chunks x 2 rows x 3 WMMAs, fully branchless.
// ---------------------------------------------------------------------------
__global__ __launch_bounds__(512) void dynconv_wmma_kernel(
    const float*  __restrict__ x,        // [B][CI][H][W]
    const __bf16* __restrict__ whi,      // [B][9][CO][CI]
    const __bf16* __restrict__ wlo,
    const float*  __restrict__ aggb,     // [B][CO]
    float*        __restrict__ out)      // [B][CO][H][W]
{
    __shared__ __bf16 sh_hi[NSTG];
    __shared__ __bf16 sh_lo[NSTG];

    const int sid    = blockIdx.x;       // 16 b * 64 row-pairs * 2 col-strips
    const int b      = sid >> 7;
    const int y0     = ((sid >> 1) & 63) << 1;
    const int xB     = (sid & 1) << 6;

    // ---------------- Phase 1: stage + split x ----------------
    const float* xb = x + (size_t)b * CI * HW;
    for (int idx = threadIdx.x; idx < NSTG; idx += 512) {
        const int col = idx % COLS;              // fastest -> coalesced global
        const int tmp = idx / COLS;
        const int row = tmp & 3;
        const int ci  = tmp >> 2;
        const int yy  = y0 - 1 + row;
        const int xx  = xB - 1 + col;
        const bool ok = (yy >= 0) && (yy < HH) && (xx >= 0) && (xx < WW);
        const float f = ok ? xb[ci * HW + yy * WW + xx] : 0.0f;
        const __bf16 h = (__bf16)f;
        sh_hi[(row * COLS + col) * CI + ci] = h;
        sh_lo[(row * COLS + col) * CI + ci] = (__bf16)(f - (float)h);
        // warm L2 for the next row-pair strip (gfx1250 global_prefetch_b8)
        if (ok && (yy + 2) < HH)
            __builtin_prefetch(&xb[ci * HW + (yy + 2) * WW + xx], 0, 1);
    }
    __syncthreads();

    // ---------------- Phase 2: WMMA ----------------
    const int lane  = threadIdx.x & 31;
    const int wave  = threadIdx.x >> 5;
    const int coT   = wave & 3;
    const int pxT   = wave >> 2;
    const int nIdx  = lane & 15;         // N (pixel) / M (C_out) within tile
    const int khalf = lane >> 4;         // K-half select (ISA A/B layout)

    const __bf16* whiB = whi + (size_t)(b * TAPS) * (CO * CI);
    const __bf16* wloB = wlo + (size_t)(b * TAPS) * (CO * CI);
    const int coA = (coT << 4) + nIdx;   // A-matrix row owned by this lane

    v8f acc0 = {}, acc1 = {};

    for (int tap = 0; tap < TAPS; ++tap) {
        const int kh = tap / 3, kw = tap % 3;
        const int colL = (pxT << 4) + nIdx + kw;          // 0..65 (halo'd)
        const __bf16* ahp = whiB + (size_t)(tap * CO + coA) * CI;
        const __bf16* alp = wloB + (size_t)(tap * CO + coA) * CI;

        #pragma unroll
        for (int hc = 0; hc < 2; ++hc) {
            const int ci0 = hc << 5;                      // ci chunk base

            // A fragments: lanes 0-15 hold K{0..15}, lanes 16-31
            // hold K{16..31} -> two 16B global loads each (L2-hot).
            Frag ah, al;
            const int k0 = ci0 + (khalf << 4);
            ah.u[0] = *(const v4u*)(ahp + k0);
            ah.u[1] = *(const v4u*)(ahp + k0 + 8);
            al.u[0] = *(const v4u*)(alp + k0);
            al.u[1] = *(const v4u*)(alp + k0 + 8);

            const int cih = ci0 + (khalf << 4);           // 16B-aligned

            #pragma unroll
            for (int r = 0; r < 2; ++r) {
                // B fragments: lane = pixel (r+kh, colL), 16 contiguous
                // bf16 along ci -> two ds_load_b128 per operand.
                const int p = ((r + kh) * COLS + colL) * CI + cih;
                Frag bh, bl;
                bh.u[0] = *(const v4u*)(sh_hi + p);
                bh.u[1] = *(const v4u*)(sh_hi + p + 8);
                bl.u[0] = *(const v4u*)(sh_lo + p);
                bl.u[1] = *(const v4u*)(sh_lo + p + 8);

                v8f a = (r == 0) ? acc0 : acc1;
                a = __builtin_amdgcn_wmma_f32_16x16x32_bf16(
                        false, ah.v, false, bh.v, (short)0, a, false, false);
                a = __builtin_amdgcn_wmma_f32_16x16x32_bf16(
                        false, ah.v, false, bl.v, (short)0, a, false, false);
                a = __builtin_amdgcn_wmma_f32_16x16x32_bf16(
                        false, al.v, false, bh.v, (short)0, a, false, false);
                if (r == 0) acc0 = a; else acc1 = a;
            }
        }
    }

    // ---- write-out: D layout VGPR v -> M = v + 8*khalf, N = lane&15 ----
    const int coBase = (coT << 4) + (khalf << 3);
    const int pxcol  = xB + (pxT << 4) + nIdx;
    const float* bp  = aggb + b * CO + coBase;
    #pragma unroll
    for (int r = 0; r < 2; ++r) {
        float* op = out + ((size_t)(b * CO + coBase)) * HW + (y0 + r) * WW + pxcol;
        const v8f a = (r == 0) ? acc0 : acc1;
        #pragma unroll
        for (int v = 0; v < 8; ++v)
            op[v * HW] = a[v] + bp[v];
    }
}

// ---------------------------------------------------------------------------
extern "C" void kernel_launch(void* const* d_in, const int* in_sizes, int n_in,
                              void* d_out, int out_size, void* d_ws, size_t ws_size,
                              hipStream_t stream) {
    const float* x       = (const float*)d_in[0];   // [16][64][128][128]
    const float* weights = (const float*)d_in[1];   // [16][5]
    const float* Wbank   = (const float*)d_in[2];   // [5][64][64][3][3]
    const float* bbank   = (const float*)d_in[3];   // [5][64]
    float*       out     = (float*)d_out;           // [16][64][128][128]

    char* ws = (char*)d_ws;
    __bf16* whi  = (__bf16*)ws;                        // 1,179,648 B
    __bf16* wlo  = (__bf16*)(ws + (size_t)NWEL * 2);   // 1,179,648 B
    float*  aggb = (float*)(ws + (size_t)NWEL * 4);    // 4,096 B

    agg_split_kernel<<<(NWEL + 255) / 256, 256, 0, stream>>>(
        weights, Wbank, bbank, whi, wlo, aggb);

    // 16 samples * 64 row-pairs * 2 col-strips = 2048 WGs of 512 threads
    dynconv_wmma_kernel<<<2048, 512, 0, stream>>>(
        x, whi, wlo, aggb, out);
}
